// Simple_TensorProduct_89979564851600
// MI455X (gfx1250) — matrix operand order repaired and checked by reading, not compiled, not run on hardware
//
#include <hip/hip_runtime.h>

typedef __attribute__((ext_vector_type(16))) _Float16 v16h;
typedef __attribute__((ext_vector_type(8)))  float    v8f;
typedef __attribute__((ext_vector_type(4)))  unsigned int uint32x4;
typedef __attribute__((ext_vector_type(8)))  int      int32x8;
typedef __attribute__((ext_vector_type(4)))  int      int32x4;

namespace {
constexpr int kMul     = 256;
constexpr int kFS      = kMul * kMul;   // 65536 floats per path
constexpr int kMZ      = 16;            // batch rows per workgroup
constexpr int kThreads = 512;           // 16 waves, one 16-col w-tile each

constexpr float kC1 = 0.04419417382415922f;   // 1/sqrt(512)
constexpr float kC2 = 0.036084391824351615f;  // 1/sqrt(768)
constexpr float kC3 = kC2;
constexpr float kC4 = 0.02551551815399144f;   // 1/sqrt(1536)
constexpr float kC5 = kC4;
}

// ---------------------------------------------------------------------------
// Weight prep: split fp32 W[p][u][w] into f16 hi/lo, rearranged into WMMA
// B-fragment order [p][ntile][kchunk][lane][h] so the GEMM inner loop's B
// load is a single aligned 32B load per lane.
// B (32x16 K-major) layout: lane&15 = N column, K = (lane>>4)*16 + h.
// ---------------------------------------------------------------------------
__global__ void tp_weight_prep(const float* __restrict__ w,
                               _Float16* __restrict__ whi,
                               _Float16* __restrict__ wlo) {
  int idx = blockIdx.x * 256 + threadIdx.x;   // 0 .. 5*16*8*32*16-1 = 327679
  int h    = idx & 15;
  int lane = (idx >> 4) & 31;
  int kc   = (idx >> 9) & 7;
  int nt   = (idx >> 12) & 15;
  int p    = idx >> 16;
  int u  = kc * 32 + (lane >> 4) * 16 + h;   // B fragment: K index
  int wc = nt * 16 + (lane & 15);            // B fragment: N index
  float x = w[p * kFS + u * kMul + wc];
  _Float16 hi = (_Float16)x;
  whi[idx] = hi;
  wlo[idx] = (_Float16)(x - (float)hi);
}

// ---------------------------------------------------------------------------
// Main kernel: 16 batch rows per block; TDM stages the raw x1 tile into LDS,
// then threads split hi/lo into WMMA A-fragment order; 16 waves each own one
// 16-wide w-tile and run the 11 shared-A split-f16 GEMM tiles + CG epilogue.
// ---------------------------------------------------------------------------
__global__ void __launch_bounds__(kThreads)
tp_main(const float* __restrict__ x1, const float* __restrict__ x2,
        const _Float16* __restrict__ whi, const _Float16* __restrict__ wlo,
        float* __restrict__ out, int n) {
  __shared__ __align__(16) float ldsRaw[kMZ][1024];         // 64 KB raw x1 tile
  // 4 A-sources (x1_0, x1_1[:,0..2]) x 8 K-chunks x 32 lanes x 16 halves
  __shared__ __align__(32) _Float16 ldsAhi[4][8][32][16];   // 32 KB
  __shared__ __align__(32) _Float16 ldsAlo[4][8][32][16];   // 32 KB

  const int tid = threadIdx.x;
  const int z0  = blockIdx.x * kMZ;

  // ---- TDM: async 2D tile load  x1[z0 : z0+16, 0:1024]  ->  ldsRaw ----
  if (tid < 32) {
    unsigned long long gaddr =
        (unsigned long long)(const void*)(x1 + (size_t)z0 * 1024);
    unsigned int ldsOff = (unsigned int)(size_t)(void*)&ldsRaw[0][0];

    uint32x4 g0;
    g0.x = 1u;                                   // count=1, is_restore=0
    g0.y = ldsOff;                               // lds_addr (bytes)
    g0.z = (unsigned int)(gaddr & 0xffffffffu);  // global_addr[31:0]
    g0.w = (unsigned int)((gaddr >> 32) & 0x01ffffffu) | (2u << 30); // type=2

    int32x8 g1;
    g1[0] = (int)(2u << 16);                 // workgroup_mask=0, data_size=4B
    g1[1] = (int)(1024u << 16);              // tensor_dim0[15:0] = 1024
    g1[2] = (int)(((unsigned)n & 0xffffu) << 16);      // tensor_dim1[15:0]
    g1[3] = (int)(((unsigned)n >> 16) | (1024u << 16)); // dim1[31:16] | tile_dim0
    g1[4] = (int)(unsigned)kMZ;              // tile_dim1=16, tile_dim2=0
    g1[5] = 1024;                            // tensor_dim0_stride[31:0]
    g1[6] = 0;                               // stride0[47:32] | stride1[15:0]
    g1[7] = 0;                               // tensor_dim1_stride[47:16]

    int32x4 gz4 = {0, 0, 0, 0};              // 2D tensor: groups 2/3 unused
    int32x8 gz8 = {0, 0, 0, 0, 0, 0, 0, 0};  // extra group (clang-23 form)

    __builtin_amdgcn_tensor_load_to_lds(g0, g1, gz4, gz4, gz8, 0);
    __builtin_amdgcn_s_wait_tensorcnt((short)0);
  }
  __syncthreads();

  // ---- split hi/lo into WMMA A-fragment layout (LDS -> LDS) ----
  // A (16x32 f16): lane&15 -> M, K=(lane>>4)*16+h.
  for (int i = 0; i < (kMZ * 1024) / kThreads; ++i) {
    int e   = i * kThreads + tid;
    int m   = e >> 10;           // batch row within tile
    int col = e & 1023;          // column of x1
    float x = ldsRaw[m][col];
    int s, u;
    if (col < kMul) { s = 0; u = col; }
    else { int cc = col - kMul; u = cc / 3; s = 1 + (cc - 3 * u); }
    int kc = u >> 5, r = u & 31;
    int hs = r >> 4;                             // lane-half select
    int h  = r & 15;                             // half index within v16h
    int ln = (hs << 4) | m;
    _Float16 hi = (_Float16)x;
    ldsAhi[s][kc][ln][h] = hi;
    ldsAlo[s][kc][ln][h] = (_Float16)(x - (float)hi);
  }
  __syncthreads();

  const int lane = tid & 31;
  const int wt   = tid >> 5;          // wave index == w-tile 0..15
  const int n0   = wt * 16;
  const int nlo  = lane & 15;
  const int mhi  = (lane >> 4) * 8;   // C/D layout: M = mhi + vgpr_index

  // per-lane x2 scalars for the 8 M rows this lane covers
  float4 x2r[8];
#pragma unroll
  for (int r = 0; r < 8; ++r) {
    int zz = z0 + mhi + r; if (zz >= n) zz = n - 1;
    x2r[r] = *(const float4*)(x2 + zz * 4);
  }

  auto x2c = [&](int r, int j) {   // x2_1[j] for row r (j is compile-time)
    return j == 0 ? x2r[r].y : (j == 1 ? x2r[r].z : x2r[r].w);
  };

  // y-tile GEMM: C += A(src s) * W(path p), K=256, split-f16 (3 wmma/chunk)
  auto runY = [&](int s, int p) {
    v8f c = {};
    const _Float16* bh = whi + (size_t)((p * 16 + wt) * 8) * 512 + lane * 16;
    const _Float16* bl = wlo + (size_t)((p * 16 + wt) * 8) * 512 + lane * 16;
#pragma unroll
    for (int kc = 0; kc < 8; ++kc) {
      v16h ah  = *(const v16h*)(&ldsAhi[s][kc][lane][0]);
      v16h al  = *(const v16h*)(&ldsAlo[s][kc][lane][0]);
      v16h vbh = *(const v16h*)(bh + kc * 512);
      v16h vbl = *(const v16h*)(bl + kc * 512);
      c = __builtin_amdgcn_wmma_f32_16x16x32_f16(false, ah, false, vbh,
                                                 (short)0, c, false, false);
      c = __builtin_amdgcn_wmma_f32_16x16x32_f16(false, ah, false, vbl,
                                                 (short)0, c, false, false);
      c = __builtin_amdgcn_wmma_f32_16x16x32_f16(false, al, false, vbh,
                                                 (short)0, c, false, false);
    }
    return c;
  };

  v8f acc0 = {};
  v8f acc1[3] = {};

  { // path0: x1_0 * W1 -> out0 (scaled by x2_0)
    v8f y = runY(0, 0);
#pragma unroll
    for (int r = 0; r < 8; ++r) acc0[r] += kC1 * x2r[r].x * y[r];
  }
  { // path1: x1_0 * W2 -> out1_k (scaled by x2_1[k])
    v8f y = runY(0, 1);
#pragma unroll
    for (int r = 0; r < 8; ++r) {
      acc1[0][r] += kC2 * x2r[r].y * y[r];
      acc1[1][r] += kC2 * x2r[r].z * y[r];
      acc1[2][r] += kC2 * x2r[r].w * y[r];
    }
  }
#pragma unroll
  for (int j = 0; j < 3; ++j) {
    { // path2: x1_1[:,j] * W3 -> out1_j (scaled by x2_0)
      v8f y = runY(1 + j, 2);
#pragma unroll
      for (int r = 0; r < 8; ++r) acc1[j][r] += kC3 * x2r[r].x * y[r];
    }
    { // path3: x1_1[:,j] * W4 -> out0 (scaled by x2_1[j])
      v8f y = runY(1 + j, 3);
#pragma unroll
      for (int r = 0; r < 8; ++r) acc0[r] += kC4 * x2c(r, j) * y[r];
    }
    { // path4: x1_1[:,j] * W5 -> cross-product epilogue
      v8f y = runY(1 + j, 4);
      const int ka = (j + 2) % 3, kb = (j + 1) % 3;
#pragma unroll
      for (int r = 0; r < 8; ++r) {
        acc1[ka][r] += kC5 * x2c(r, (j + 1) % 3) * y[r];
        acc1[kb][r] -= kC5 * x2c(r, (j + 2) % 3) * y[r];
      }
    }
  }

  // ---- store: out[:, :256]=out0 ; out[:, 256 + w*3 + k]=out1 ----
#pragma unroll
  for (int r = 0; r < 8; ++r) {
    int z = z0 + mhi + r;
    if (z < n) {
      int w = n0 + nlo;
      out[z * 1024 + w] = acc0[r];
      float* o1 = out + z * 1024 + 256 + w * 3;
      o1[0] = acc1[0][r];
      o1[1] = acc1[1][r];
      o1[2] = acc1[2][r];
    }
  }
}

// ---------------------------------------------------------------------------
extern "C" void kernel_launch(void* const* d_in, const int* in_sizes, int n_in,
                              void* d_out, int out_size, void* d_ws, size_t ws_size,
                              hipStream_t stream) {
  (void)n_in; (void)out_size; (void)ws_size;
  const float* x1 = (const float*)d_in[0];
  const float* x2 = (const float*)d_in[1];
  const float* w  = (const float*)d_in[2];
  float* out = (float*)d_out;

  _Float16* whi = (_Float16*)d_ws;          // 5*65536 halves
  _Float16* wlo = whi + 5 * kFS;            // 5*65536 halves (1.31 MB total)

  int n = in_sizes[0] / 1024;               // 20000

  tp_weight_prep<<<(5 * kFS) / 256, 256, 0, stream>>>(w, whi, wlo);
  tp_main<<<(n + kMZ - 1) / kMZ, kThreads, 0, stream>>>(x1, x2, whi, wlo, out, n);
}
